// MixingBlock_58995670778482
// MI455X (gfx1250) — matrix operand order repaired and checked by reading, not compiled, not run on hardware
//
#include <hip/hip_runtime.h>

typedef __bf16 bf16;
typedef __attribute__((ext_vector_type(16))) __bf16 v16bf;
typedef __attribute__((ext_vector_type(8)))  float  v8f;

#define B_DIM 8
#define S_DIM 4096
#define FEAT  480
#define NCH   224

static __device__ __forceinline__ v8f v8zero(){
  v8f z = {0.f,0.f,0.f,0.f,0.f,0.f,0.f,0.f};
  return z;
}

static __device__ __forceinline__ v8f wmma_bf(v16bf a, v16bf b, v8f c){
  // D = A(16x32 bf16) * B(32x16 bf16) + C(f32)
  return __builtin_amdgcn_wmma_f32_16x16x32_bf16(false, a, false, b, (short)0, c, false, false);
}

// Packed A-fragments in global ws: frag = mt*KT+kt, 512 bf16 each, lane-major (lane*16 + e)
static __device__ __forceinline__ v16bf load_a_frag(const bf16* __restrict__ pw, int frag, int lane){
  return *reinterpret_cast<const v16bf*>(pw + (size_t)frag*512 + (size_t)lane*16);
}

// B fragment from LDS in fragment-native layout buf[(col)*K + k]:
// column n = token*DC + d  (token shifted by coloff), k contiguous -> one aligned 32B chunk per lane.
template<int DC, int K>
static __device__ __forceinline__ v16bf load_bfrag(const bf16* __restrict__ buf, int nt, int kt,
                                                   int lane, int coloff){
  const int n = nt*16 + (lane & 15);
  const int t = n / DC;
  const int d = n - t*DC;
  const int kadd = (lane < 16) ? 0 : 16;
  return *reinterpret_cast<const v16bf*>(buf + ((t + coloff)*DC + d)*K + kt*32 + kadd);
}

// ---------------- weight pre-pack into WMMA A-fragment layout (f32 -> bf16) ----------------
__global__ void pack_w(const float* __restrict__ src, bf16* __restrict__ dst, int M, int K, int trans){
  const int Kt = K >> 5;
  const int total = (M >> 4) * Kt * 512;
  for (int i = blockIdx.x*blockDim.x + threadIdx.x; i < total; i += gridDim.x*blockDim.x){
    const int e    = i & 15;
    const int lane = (i >> 4) & 31;
    const int frag = i >> 9;
    const int kt = frag % Kt;
    const int mt = frag / Kt;
    const int row = mt*16 + (lane & 15);
    const int kk  = e + ((lane < 16) ? 0 : 16);   // A-matrix 16x32 bf16 layout, k contiguous: k = 16*half + e
    const int col = kt*32 + kk;
    const float v = trans ? src[(size_t)col*M + row] : src[(size_t)row*K + col];
    dst[i] = (bf16)v;
  }
}

// ---------------- kernel A: channel mixing + equivariant layernorm ----------------
template<int MT,int KT,int NT,int DC,int K,int FBASE,int GBASE>
static __device__ void mix_part(const bf16* __restrict__ sB,
                                const bf16* __restrict__ pw,
                                const float* __restrict__ gamma,
                                const float* __restrict__ beta,
                                bf16* __restrict__ sO,
                                float* colsq, int lane)
{
  v8f acc[MT*NT];
#pragma unroll
  for (int i=0;i<MT*NT;++i) acc[i] = v8zero();
#pragma unroll
  for (int kt=0; kt<KT; ++kt){
    v16bf b[NT];
#pragma unroll
    for (int nt=0; nt<NT; ++nt) b[nt] = load_bfrag<DC,K>(sB, nt, kt, lane, 0);
#pragma unroll
    for (int mt=0; mt<MT; ++mt){
      v16bf a = load_a_frag(pw, mt*KT + kt, lane);
#pragma unroll
      for (int nt=0; nt<NT; ++nt) acc[mt*NT + nt] = wmma_bf(a, b[nt], acc[mt*NT + nt]);
    }
  }
  const int hi  = (lane >> 4) & 1;
  const int col = lane & 15;
  if (DC == 1){
    float s1 = 0.f, s2 = 0.f;
#pragma unroll
    for (int mt=0; mt<MT; ++mt)
#pragma unroll
      for (int e=0;e<8;++e){ float v = acc[mt][e]; s1 += v; s2 += v*v; }
    s1 += __shfl_xor(s1, 16, 32);
    s2 += __shfl_xor(s2, 16, 32);
    const float mu  = s1 / (float)(MT*16);
    const float var = s2 / (float)(MT*16) - mu*mu;
    const float rs  = rsqrtf(var + 1e-6f);
#pragma unroll
    for (int mt=0; mt<MT; ++mt)
#pragma unroll
      for (int e=0;e<8;++e){
        const int o = mt*16 + hi*8 + e;
        const float v = (acc[mt][e] - mu)*rs*gamma[GBASE+o] + beta[GBASE+o];
        sO[col*FEAT + FBASE + o] = (bf16)v;
      }
  } else {
    __syncthreads();
#pragma unroll
    for (int nt=0; nt<NT; ++nt){
      float s2 = 0.f;
#pragma unroll
      for (int mt=0; mt<MT; ++mt)
#pragma unroll
        for (int e=0;e<8;++e){ float v = acc[mt*NT+nt][e]; s2 += v*v; }
      s2 += __shfl_xor(s2, 16, 32);
      if (lane < 16) colsq[nt*16 + col] = s2;
    }
    __syncthreads();
#pragma unroll
    for (int nt=0; nt<NT; ++nt){
      const int n = nt*16 + col;
      const int t = n / DC;
      const int d = n - t*DC;
      float ssum = 0.f;
#pragma unroll
      for (int dd=0; dd<DC; ++dd) ssum += colsq[t*DC + dd];
      const float rinv = rsqrtf(ssum / (float)(MT*16) + 1e-6f);
#pragma unroll
      for (int mt=0; mt<MT; ++mt)
#pragma unroll
        for (int e=0;e<8;++e){
          const int o = mt*16 + hi*8 + e;
          sO[t*FEAT + FBASE + o*DC + d] = (bf16)(acc[mt*NT+nt][e] * rinv * gamma[GBASE+o]);
        }
    }
  }
}

__global__ __launch_bounds__(32)
void k_mix_ln(const float* __restrict__ x,
              const bf16*  __restrict__ pw,
              const float* __restrict__ gamma,
              const float* __restrict__ beta,
              bf16* __restrict__ hln)
{
  __shared__ alignas(32) bf16 sB0[16*128];   // [t][c]          (l=0)
  __shared__ alignas(32) bf16 sB1[48*64];    // [t*3+d][m]      (l=1)
  __shared__ alignas(32) bf16 sB2[80*32];    // [t*5+d][m]      (l=2)
  __shared__ alignas(32) bf16 sO[16*FEAT];
  __shared__ float colsq[96];
  const int lane = threadIdx.x;
  const long tok0 = (long)blockIdx.x * 16;
  {
    const int t = lane & 15, half = lane >> 4;
    const float* src = x + (tok0 + t)*FEAT + half*240;
#pragma unroll 4
    for (int i=0;i<240;++i){
      const int c = half*240 + i;
      const bf16 v = (bf16)src[i];
      if (c < 128)      sB0[t*128 + c] = v;
      else if (c < 320){ const int cc = c-128, m = cc/3, d = cc - 3*m; sB1[(t*3+d)*64 + m] = v; }
      else             { const int cc = c-320, m = cc/5, d = cc - 5*m; sB2[(t*5+d)*32 + m] = v; }
    }
  }
  __syncthreads();
  mix_part<8,4,1,1,128,  0,  0>(sB0, pw +     0, gamma, beta, sO, colsq, lane);  // l=0
  mix_part<4,2,3,3, 64,128,128>(sB1, pw + 16384, gamma, beta, sO, colsq, lane);  // l=1
  mix_part<2,1,5,5, 32,320,192>(sB2, pw + 20480, gamma, beta, sO, colsq, lane);  // l=2
  __syncthreads();
  {
    const int t = lane & 15, half = lane >> 4;
    bf16* dst = hln + (tok0 + t)*FEAT + half*240;
#pragma unroll 4
    for (int i=0;i<240;++i) dst[i] = sO[t*FEAT + half*240 + i];
  }
}

// ---------------- kernel B: 3-offset neighbor conv + residual + final layernorm ----------------
template<int MT,int KT,int NT,int DC,int K,int FBASE,int GBASE,bool VEC>
static __device__ void conv_part(const bf16* __restrict__ sB,    // part-specific neighbor buffer
                                 const bf16* __restrict__ sH1,   // l=1 buffer (for cross term)
                                 const bf16* __restrict__ pwc, int matStride,
                                 const bf16* __restrict__ radL,
                                 const bf16* __restrict__ svL,
                                 const float* __restrict__ uL,
                                 const float* __restrict__ maskL,
                                 const float* __restrict__ wvv,
                                 const float* __restrict__ gamma,
                                 const float* __restrict__ beta,
                                 const float* __restrict__ x,
                                 float* __restrict__ out,
                                 long rowbase, float* colsq, int lane)
{
  v8f acc[MT*NT];
#pragma unroll
  for (int i=0;i<MT*NT;++i) acc[i] = v8zero();
  const int hi  = (lane >> 4) & 1;
  const int col = lane & 15;
#pragma unroll
  for (int ko=0; ko<3; ++ko){
    v8f dk[MT*NT];
#pragma unroll
    for (int i=0;i<MT*NT;++i) dk[i] = v8zero();
#pragma unroll
    for (int kt=0; kt<KT; ++kt){
      v16bf b[NT];
#pragma unroll
      for (int nt=0; nt<NT; ++nt) b[nt] = load_bfrag<DC,K>(sB, nt, kt, lane, ko);
#pragma unroll
      for (int mt=0; mt<MT; ++mt){
        v16bf a = load_a_frag(pwc + (size_t)ko*matStride, mt*KT + kt, lane);
#pragma unroll
        for (int nt=0; nt<NT; ++nt) dk[mt*NT + nt] = wmma_bf(a, b[nt], dk[mt*NT + nt]);
      }
    }
#pragma unroll
    for (int nt=0; nt<NT; ++nt){
      const int n = nt*16 + col;
      const int t = n / DC;
      const int d = n - t*DC;
      const float m = maskL[t*3 + ko];
#pragma unroll
      for (int mt=0; mt<MT; ++mt){
#pragma unroll
        for (int e=0;e<8;++e){
          const int o = mt*16 + hi*8 + e;
          float v = m * (float)radL[(ko*NCH + GBASE + o)*16 + t] * dk[mt*NT + nt][e];
          if (VEC){
            v += m * (float)svL[(ko*64 + o)*16 + t] * uL[(t*3 + ko)*3 + d];
            int d1 = d + 1; if (d1 >= 3) d1 -= 3;
            int d2 = d + 2; if (d2 >= 3) d2 -= 3;
            const float n1 = (float)sH1[((t + ko)*3 + d1)*64 + o];
            const float n2 = (float)sH1[((t + ko)*3 + d2)*64 + o];
            const float u1 = uL[(t*3 + ko)*3 + d1];
            const float u2 = uL[(t*3 + ko)*3 + d2];
            v += m * wvv[ko*64 + o] * (n1*u2 - n2*u1);   // cross(nb1,u)[d]
          }
          acc[mt*NT + nt][e] += v;
        }
      }
    }
  }
  // residual
#pragma unroll
  for (int nt=0; nt<NT; ++nt){
    const int n = nt*16 + col;
    const int t = n / DC;
    const int d = n - t*DC;
    const float* xr = x + (rowbase + t)*FEAT;
#pragma unroll
    for (int mt=0; mt<MT; ++mt)
#pragma unroll
      for (int e=0;e<8;++e){
        const int o = mt*16 + hi*8 + e;
        acc[mt*NT + nt][e] += xr[FBASE + o*DC + d];
      }
  }
  // final equivariant layernorm + global store
  if (DC == 1){
    float s1 = 0.f, s2 = 0.f;
#pragma unroll
    for (int mt=0; mt<MT; ++mt)
#pragma unroll
      for (int e=0;e<8;++e){ float v = acc[mt][e]; s1 += v; s2 += v*v; }
    s1 += __shfl_xor(s1, 16, 32);
    s2 += __shfl_xor(s2, 16, 32);
    const float mu  = s1 / (float)(MT*16);
    const float var = s2 / (float)(MT*16) - mu*mu;
    const float rs  = rsqrtf(var + 1e-6f);
    float* orow = out + (rowbase + col)*FEAT;
#pragma unroll
    for (int mt=0; mt<MT; ++mt)
#pragma unroll
      for (int e=0;e<8;++e){
        const int o = mt*16 + hi*8 + e;
        orow[FBASE + o] = (acc[mt][e] - mu)*rs*gamma[GBASE+o] + beta[GBASE+o];
      }
  } else {
    __syncthreads();
#pragma unroll
    for (int nt=0; nt<NT; ++nt){
      float s2 = 0.f;
#pragma unroll
      for (int mt=0; mt<MT; ++mt)
#pragma unroll
        for (int e=0;e<8;++e){ float v = acc[mt*NT+nt][e]; s2 += v*v; }
      s2 += __shfl_xor(s2, 16, 32);
      if (lane < 16) colsq[nt*16 + col] = s2;
    }
    __syncthreads();
#pragma unroll
    for (int nt=0; nt<NT; ++nt){
      const int n = nt*16 + col;
      const int t = n / DC;
      const int d = n - t*DC;
      float ssum = 0.f;
#pragma unroll
      for (int dd=0; dd<DC; ++dd) ssum += colsq[t*DC + dd];
      const float rinv = rsqrtf(ssum / (float)(MT*16) + 1e-6f);
      float* orow = out + (rowbase + t)*FEAT;
#pragma unroll
      for (int mt=0; mt<MT; ++mt)
#pragma unroll
        for (int e=0;e<8;++e){
          const int o = mt*16 + hi*8 + e;
          orow[FBASE + o*DC + d] = acc[mt*NT+nt][e] * rinv * gamma[GBASE+o];
        }
    }
  }
}

__global__ __launch_bounds__(32)
void k_conv(const float* __restrict__ x, const float* __restrict__ coords,
            const bf16*  __restrict__ hln, const bf16* __restrict__ pw,
            const float* __restrict__ Wr1, const float* __restrict__ br1,
            const float* __restrict__ br2, const float* __restrict__ wvv,
            const float* __restrict__ gamma, const float* __restrict__ beta,
            float* __restrict__ out)
{
  __shared__ alignas(32) bf16 sH0[18*128];      // [lt][c]       neighbor h_ln l=0
  __shared__ alignas(32) bf16 sH1[18*3*64];     // [lt*3+d][m]   neighbor h_ln l=1
  __shared__ alignas(32) bf16 sH2[18*5*32];     // [lt*5+d][m]   neighbor h_ln l=2
  __shared__ bf16  radL[3*NCH*16];              // radial[ko][chan][t]
  __shared__ bf16  svL[3*64*16];                // sv[ko][o][t]
  __shared__ float uL[48*3];
  __shared__ float maskL[48];
  __shared__ float rL[48];
  __shared__ float w1L[192];
  __shared__ float b1L[192];
  __shared__ float colsq[96];
  const int lane = threadIdx.x;
  const int b  = blockIdx.x / (S_DIM/16);
  const int s0 = (blockIdx.x % (S_DIM/16)) * 16;
  const long rowbase = (long)b*S_DIM + s0;

  // stage neighbor h_ln (tokens s0-1 .. s0+16) into fragment-native part buffers
  for (int lt=0; lt<18; ++lt){
    int s = s0 + lt - 1;
    s = s < 0 ? 0 : (s >= S_DIM ? S_DIM-1 : s);
    const bf16* src = hln + ((long)b*S_DIM + s)*FEAT;
    for (int c = lane; c < FEAT; c += 32){
      const bf16 v = src[c];
      if (c < 128)      sH0[lt*128 + c] = v;
      else if (c < 320){ const int cc = c-128, m = cc/3, d = cc - 3*m; sH1[(lt*3+d)*64 + m] = v; }
      else             { const int cc = c-320, m = cc/5, d = cc - 5*m; sH2[(lt*5+d)*32 + m] = v; }
    }
  }
  // radial-MLP layer-1 weights + rel geometry
  for (int i = lane; i < 192; i += 32){ w1L[i] = Wr1[i]; b1L[i] = br1[i]; }
  for (int i = lane; i < 48; i += 32){
    const int t = i/3, ko = i - t*3, off = ko - 1;
    const int s = s0 + t, sn = s + off;
    const float m = (sn >= 0 && sn < S_DIM) ? 1.f : 0.f;
    const int snc = sn < 0 ? 0 : (sn >= S_DIM ? S_DIM-1 : sn);
    const float rx = coords[((long)b*S_DIM+snc)*3+0] - coords[((long)b*S_DIM+s)*3+0];
    const float ry = coords[((long)b*S_DIM+snc)*3+1] - coords[((long)b*S_DIM+s)*3+1];
    const float rz = coords[((long)b*S_DIM+snc)*3+2] - coords[((long)b*S_DIM+s)*3+2];
    const float r = sqrtf(rx*rx + ry*ry + rz*rz + 1e-12f);
    rL[i] = r; maskL[i] = m;
    uL[i*3+0] = rx/r; uL[i*3+1] = ry/r; uL[i*3+2] = rz/r;
  }
  __syncthreads();

  // radial MLP output layer via WMMA: rad = silu(r*W1+b1) @ Wr2 + br2   (M=224,K=64,N=16 per offset)
  const bf16* pWr2T = pw + 110592;
  const int hi = (lane >> 4) & 1, tcol = lane & 15;
  for (int ko=0; ko<3; ++ko){
    v8f racc[14];
#pragma unroll
    for (int i=0;i<14;++i) racc[i] = v8zero();
#pragma unroll
    for (int kt=0; kt<2; ++kt){
      const int kadd = (lane < 16) ? 0 : 16;
      const float rv = rL[tcol*3 + ko];
      v16bf bfrag;
#pragma unroll
      for (int e=0;e<16;++e){
        const int h = kt*32 + kadd + e;
        const float z = rv * w1L[ko*64 + h] + b1L[ko*64 + h];
        bfrag[e] = (bf16)(z / (1.f + expf(-z)));
      }
#pragma unroll
      for (int mt=0; mt<14; ++mt){
        v16bf a = load_a_frag(pWr2T + (size_t)ko*14336, mt*2 + kt, lane);
        racc[mt] = wmma_bf(a, bfrag, racc[mt]);
      }
    }
#pragma unroll
    for (int mt=0; mt<14; ++mt)
#pragma unroll
      for (int e=0;e<8;++e){
        const int o = mt*16 + hi*8 + e;
        radL[(ko*NCH + o)*16 + tcol] = (bf16)(racc[mt][e] + br2[ko*NCH + o]);
      }
  }
  // sv = Wsv[k] @ nb0   (M=64,K=128,N=16 per offset)
  const bf16* pWsv = pw + 86016;
  for (int ko=0; ko<3; ++ko){
    v8f sacc[4];
#pragma unroll
    for (int i=0;i<4;++i) sacc[i] = v8zero();
#pragma unroll
    for (int kt=0; kt<4; ++kt){
      v16bf bfrag = load_bfrag<1,128>(sH0, 0, kt, lane, ko);
#pragma unroll
      for (int mt=0; mt<4; ++mt){
        v16bf a = load_a_frag(pWsv + (size_t)ko*8192, mt*4 + kt, lane);
        sacc[mt] = wmma_bf(a, bfrag, sacc[mt]);
      }
    }
#pragma unroll
    for (int mt=0; mt<4; ++mt)
#pragma unroll
      for (int e=0;e<8;++e){
        const int o = mt*16 + hi*8 + e;
        svL[(ko*64 + o)*16 + tcol] = (bf16)sacc[mt][e];
      }
  }
  __syncthreads();

  conv_part<8,4,1,1,128,  0,  0,false>(sH0, sH1, pw + 21504, 16384, radL, svL, uL, maskL, wvv, gamma, beta, x, out, rowbase, colsq, lane);
  conv_part<4,2,3,3, 64,128,128,true >(sH1, sH1, pw + 70656,  4096, radL, svL, uL, maskL, wvv, gamma, beta, x, out, rowbase, colsq, lane);
  conv_part<2,1,5,5, 32,320,192,false>(sH2, sH1, pw + 82944,  1024, radL, svL, uL, maskL, wvv, gamma, beta, x, out, rowbase, colsq, lane);
}

// ---------------- host launch ----------------
extern "C" void kernel_launch(void* const* d_in, const int* in_sizes, int n_in,
                              void* d_out, int out_size, void* d_ws, size_t ws_size,
                              hipStream_t stream)
{
  const float* x        = (const float*)d_in[0];
  const float* coords   = (const float*)d_in[1];
  const float* W0       = (const float*)d_in[2];
  const float* W1       = (const float*)d_in[3];
  const float* W2       = (const float*)d_in[4];
  const float* gamma_si = (const float*)d_in[5];
  const float* beta_si  = (const float*)d_in[6];
  const float* Wc0      = (const float*)d_in[7];
  const float* Wc1      = (const float*)d_in[8];
  const float* Wc2      = (const float*)d_in[9];
  const float* Wsv      = (const float*)d_in[10];
  const float* wvv      = (const float*)d_in[11];
  const float* Wr1      = (const float*)d_in[12];
  const float* br1      = (const float*)d_in[13];
  const float* Wr2      = (const float*)d_in[14];
  const float* br2      = (const float*)d_in[15];
  const float* gamma_ln = (const float*)d_in[16];
  const float* beta_ln  = (const float*)d_in[17];
  float* out = (float*)d_out;

  bf16* pw  = (bf16*)d_ws;                              // packed weights: 153600 bf16
  bf16* hln = (bf16*)((char*)d_ws + (1u << 19));        // h_ln: 32768*480 bf16

  auto pk = [&](const float* src, size_t off, int M, int K, int trans){
    const int total = (M/16)*(K/32)*512;
    pack_w<<<(total + 255)/256, 256, 0, stream>>>(src, pw + off, M, K, trans);
  };
  pk(W0,      0, 128, 128, 0);
  pk(W1,  16384,  64,  64, 0);
  pk(W2,  20480,  32,  32, 0);
  for (int k = 0; k < 3; ++k){
    pk(Wc0 + (size_t)k*16384, 21504 + (size_t)k*16384, 128, 128, 0);
    pk(Wc1 + (size_t)k*4096,  70656 + (size_t)k*4096,   64,  64, 0);
    pk(Wc2 + (size_t)k*1024,  82944 + (size_t)k*1024,   32,  32, 0);
    pk(Wsv + (size_t)k*8192,  86016 + (size_t)k*8192,   64, 128, 0);
    pk(Wr2 + (size_t)k*14336, 110592 + (size_t)k*14336, 224,  64, 1);  // transposed: A[c][h]=Wr2[h][c]
  }

  const int nblocks = (B_DIM*S_DIM)/16;   // 2048 waves of 16 tokens
  k_mix_ln<<<nblocks, 32, 0, stream>>>(x, pw, gamma_si, beta_si, hln);
  k_conv  <<<nblocks, 32, 0, stream>>>(x, coords, hln, pw, Wr1, br1, br2, wvv, gamma_ln, beta_ln, out);

  (void)in_sizes; (void)n_in; (void)out_size; (void)ws_size;
}
